// MultiHeadAttention_14018773254670
// MI455X (gfx1250) — compile-verified, matrix operand order checked by reading, not run on hardware
//
#include <hip/hip_runtime.h>

typedef __attribute__((ext_vector_type(16))) _Float16 v16h;
typedef __attribute__((ext_vector_type(8)))  _Float16 v8h;
typedef __attribute__((ext_vector_type(2)))  _Float16 v2h;
typedef __attribute__((ext_vector_type(8)))  float    v8f;
typedef __attribute__((ext_vector_type(4)))  int      v4i;

#define D_MODEL 1024
#define SEQ     2048
#define BATCH   4
#define HEADS   16
#define HDIM    64

// Padded LDS row strides (halfs). 40 halfs = 80B rows -> lane dword-stride 20,
// 16 fragment lanes land on 16 distinct banks; 72 halfs = 144B for the 64-wide
// K rows. All fragment offsets stay 16B-aligned for ds_load_b128.
#define AS_STRIDE 40
#define BS_STRIDE 40
#define KS_STRIDE 72
#define VT_STRIDE 40

static __device__ __forceinline__ v16h cat8(v8h lo, v8h hi) {
  return __builtin_shufflevector(lo, hi, 0,1,2,3,4,5,6,7,8,9,10,11,12,13,14,15);
}
static __device__ __forceinline__ v8f vzero8() {
  v8f z;
#pragma unroll
  for (int i = 0; i < 8; ++i) z[i] = 0.0f;
  return z;
}
static __device__ __forceinline__ v8f wmma_f16(v16h a, v16h b, v8f c) {
  return __builtin_amdgcn_wmma_f32_16x16x32_f16(false, a, false, b, (short)0, c,
                                                false, false);
}

// ---- optional gfx1250 async global->LDS copy (ASYNCcnt path) --------------
#if defined(__has_builtin)
#if __has_builtin(__builtin_amdgcn_global_load_async_to_lds_b128)
#define HAVE_ASYNC_LDS 1
#endif
#endif

#ifdef HAVE_ASYNC_LDS
typedef __attribute__((address_space(1))) v4i* gv4p;
typedef __attribute__((address_space(3))) v4i* lv4p;
static __device__ __forceinline__ void async_cp16(const void* g, void* l) {
  __builtin_amdgcn_global_load_async_to_lds_b128((gv4p)g, (lv4p)l, 0, 0);
}
static __device__ __forceinline__ void wait_async0() {
#if __has_builtin(__builtin_amdgcn_s_wait_asynccnt)
  __builtin_amdgcn_s_wait_asynccnt(0);
#else
  asm volatile("s_wait_asynccnt 0x0" ::: "memory");
#endif
}
#endif

// ---------------------------------------------------------------------------
// Kernel 1: fused QKV projection. Block tile 128(M) x 128(N), BK=32,
// 256 threads = 8 waves in 4(M)x2(N); wave tile 32x64 -> 8 WMMA per k-step.
// Register-prefetch + double-buffered LDS: 1 barrier per k-step.
// Output f16 in [B, H, T, Dh]; Q pre-scaled by 1/sqrt(Dh).
// ---------------------------------------------------------------------------
__global__ __launch_bounds__(256)
void qkv_proj_kernel(const float* __restrict__ x,
                     const float* __restrict__ Wq,
                     const float* __restrict__ Wk,
                     const float* __restrict__ Wv,
                     _Float16* __restrict__ Qh,
                     _Float16* __restrict__ Kh,
                     _Float16* __restrict__ Vh) {
  __shared__ _Float16 As[2][128 * AS_STRIDE];  // [row][k]
  __shared__ _Float16 Bs[2][128 * BS_STRIDE];  // transposed: [n][k]

  const int tid  = threadIdx.x;
  const int wave = tid >> 5;
  const int lane = tid & 31;
  const int l    = lane & 15;
  const int hi   = lane >> 4;

  const int m0      = blockIdx.x * 128;
  const int nGlobal = blockIdx.y * 128;
  const int mat     = nGlobal >> 10;  // 0=Q 1=K 2=V (128 | 1024)
  const int n0      = nGlobal & 1023;
  const float* W    = (mat == 0) ? Wq : (mat == 1) ? Wk : Wv;
  _Float16* Dst     = (mat == 0) ? Qh : (mat == 1) ? Kh : Vh;
  const float cScl  = (mat == 0) ? 0.125f : 1.0f;  // 1/sqrt(64) on Q

  // staging assignments
  const int aRow = tid >> 1;          // 0..127
  const int aK   = (tid & 1) * 16;    // 0,16
  const int bKp  = (tid >> 4) * 2;    // 0..30 (even k pair)
  const int bN   = (tid & 15) * 8;    // 0..120

  v8h ra0, ra1;   // A tile regs (16 halfs)
  v2h rb[8];      // B tile regs (pairs along k)

  auto gload = [&](int k0) {
    const float4* asrc =
        (const float4*)(x + (size_t)(m0 + aRow) * D_MODEL + k0 + aK);
    float4 f0 = asrc[0], f1 = asrc[1], f2 = asrc[2], f3 = asrc[3];
    ra0[0]=(_Float16)f0.x; ra0[1]=(_Float16)f0.y; ra0[2]=(_Float16)f0.z; ra0[3]=(_Float16)f0.w;
    ra0[4]=(_Float16)f1.x; ra0[5]=(_Float16)f1.y; ra0[6]=(_Float16)f1.z; ra0[7]=(_Float16)f1.w;
    ra1[0]=(_Float16)f2.x; ra1[1]=(_Float16)f2.y; ra1[2]=(_Float16)f2.z; ra1[3]=(_Float16)f2.w;
    ra1[4]=(_Float16)f3.x; ra1[5]=(_Float16)f3.y; ra1[6]=(_Float16)f3.z; ra1[7]=(_Float16)f3.w;

    const float4* b0 =
        (const float4*)(W + (size_t)(k0 + bKp) * D_MODEL + n0 + bN);
    const float4* b1 =
        (const float4*)(W + (size_t)(k0 + bKp + 1) * D_MODEL + n0 + bN);
    float4 p0 = b0[0], p1 = b0[1];
    float4 q0 = b1[0], q1 = b1[1];
    float r0[8] = {p0.x, p0.y, p0.z, p0.w, p1.x, p1.y, p1.z, p1.w};
    float r1[8] = {q0.x, q0.y, q0.z, q0.w, q1.x, q1.y, q1.z, q1.w};
#pragma unroll
    for (int j = 0; j < 8; ++j) {
      rb[j][0] = (_Float16)r0[j];
      rb[j][1] = (_Float16)r1[j];
    }
  };
  auto lstore = [&](int buf) {
    *(v8h*)&As[buf][aRow * AS_STRIDE + aK]     = ra0;
    *(v8h*)&As[buf][aRow * AS_STRIDE + aK + 8] = ra1;
#pragma unroll
    for (int j = 0; j < 8; ++j)
      *(v2h*)&Bs[buf][(bN + j) * BS_STRIDE + bKp] = rb[j];
  };

  v8f acc[2][4];
#pragma unroll
  for (int s = 0; s < 2; ++s)
#pragma unroll
    for (int i = 0; i < 4; ++i) acc[s][i] = vzero8();

  const int mb = (wave & 3) * 32;   // wave row base within block
  const int nb = (wave >> 2) * 64;  // wave col base within block

  gload(0);
  lstore(0);
  for (int kt = 0; kt < 32; ++kt) {
    const int cur = kt & 1;
    if (kt + 1 < 32) gload((kt + 1) * 32);  // issue next global loads early
    __syncthreads();

    const int r0 = (mb + l) * AS_STRIDE;
    const int r1 = (mb + 16 + l) * AS_STRIDE;
    v16h a0 = cat8(*(const v8h*)&As[cur][r0 + hi * 8],
                   *(const v8h*)&As[cur][r0 + 16 + hi * 8]);
    v16h a1 = cat8(*(const v8h*)&As[cur][r1 + hi * 8],
                   *(const v8h*)&As[cur][r1 + 16 + hi * 8]);
#pragma unroll
    for (int i = 0; i < 4; ++i) {
      const int nr = (nb + i * 16 + l) * BS_STRIDE;
      v16h b = cat8(*(const v8h*)&Bs[cur][nr + hi * 8],
                    *(const v8h*)&Bs[cur][nr + 16 + hi * 8]);
      acc[0][i] = wmma_f16(a0, b, acc[0][i]);
      acc[1][i] = wmma_f16(a1, b, acc[1][i]);
    }
    if (kt + 1 < 32) lstore(cur ^ 1);
  }

  // store to [B, H, T, Dh] f16
#pragma unroll
  for (int s = 0; s < 2; ++s) {
#pragma unroll
    for (int i = 0; i < 4; ++i) {
      const int c  = n0 + nb + i * 16 + l;
      const int h  = c >> 6;
      const int dd = c & 63;
#pragma unroll
      for (int r = 0; r < 8; ++r) {
        const int m = m0 + mb + s * 16 + hi * 8 + r;
        const int b = m >> 11;
        const int t = m & 2047;
        Dst[(((size_t)(b * HEADS + h)) * SEQ + t) * HDIM + dd] =
            (_Float16)(acc[s][i][r] * cScl);
      }
    }
  }
}

// ---------------------------------------------------------------------------
// Kernel 2: causal flash attention, transposed-MMA formulation.
// Block = 128 threads (4 waves); wave owns 16 queries of one (b,h).
// S^T = K.Q^T and O^T = V^T.P^T so softmax + P pack stay in-lane.
// ---------------------------------------------------------------------------
__global__ __launch_bounds__(128)
void attention_kernel(const _Float16* __restrict__ Qh,
                      const _Float16* __restrict__ Kh,
                      const _Float16* __restrict__ Vh,
                      _Float16* __restrict__ Ctx) {
  __shared__ _Float16 Ks[32 * KS_STRIDE];  // [key][d], padded
  __shared__ _Float16 Vt[64 * VT_STRIDE];  // transposed [d][key], padded

  const int tid  = threadIdx.x;
  const int wave = tid >> 5;
  const int lane = tid & 31;
  const int l    = lane & 15;
  const int hi   = lane >> 4;

  const int q0blk   = blockIdx.x * 64;
  const int bh      = blockIdx.y;
  const size_t base = (size_t)bh * SEQ * HDIM;

  const int q0w = q0blk + wave * 16;
  const int qg  = q0w + l;  // lanes l and l+16 share this query

  // Q as B-operand fragments, loaded once (2 frags cover Dh=64)
  v16h qb[2];
  {
    const _Float16* qrow = Qh + base + (size_t)(q0w + l) * HDIM;
#pragma unroll
    for (int f = 0; f < 2; ++f) {
      qb[f] = cat8(*(const v8h*)(qrow + f * 32 + hi * 8),
                   *(const v8h*)(qrow + f * 32 + 16 + hi * 8));
    }
  }

  v8f o[4];
#pragma unroll
  for (int i = 0; i < 4; ++i) o[i] = vzero8();
  float mrow = -3.0e38f;
  float lrow = 0.0f;

  const int sKey = tid >> 2;        // 0..31
  const int sD   = (tid & 3) * 16;  // 0,16,32,48

  const int kbEnd = (q0blk + 63) >> 5;
  for (int kb = 0; kb <= kbEnd; ++kb) {
    const int key0 = kb * 32;
    // --- stage K rows (async path when available) and V transposed ---
    {
      const _Float16* ksrc = Kh + base + (size_t)(key0 + sKey) * HDIM + sD;
#ifdef HAVE_ASYNC_LDS
      async_cp16(ksrc,     &Ks[sKey * KS_STRIDE + sD]);
      async_cp16(ksrc + 8, &Ks[sKey * KS_STRIDE + sD + 8]);
#else
      *(v8h*)&Ks[sKey * KS_STRIDE + sD]     = *(const v8h*)(ksrc);
      *(v8h*)&Ks[sKey * KS_STRIDE + sD + 8] = *(const v8h*)(ksrc + 8);
#endif
      const _Float16* vsrc = Vh + base + (size_t)(key0 + sKey) * HDIM + sD;
      v8h v0 = *(const v8h*)(vsrc);
      v8h v1 = *(const v8h*)(vsrc + 8);
#pragma unroll
      for (int j = 0; j < 8; ++j) {
        Vt[(sD + j) * VT_STRIDE + sKey]       = v0[j];
        Vt[(sD + 8 + j) * VT_STRIDE + sKey]   = v1[j];
      }
#ifdef HAVE_ASYNC_LDS
      wait_async0();
#endif
    }
    __syncthreads();

    if (key0 <= q0w + 15) {  // wave still has causal work in this block
      v8f st[2];
#pragma unroll
      for (int kt = 0; kt < 2; ++kt) {
        st[kt] = vzero8();
        const int kr = (kt * 16 + l) * KS_STRIDE;
#pragma unroll
        for (int f = 0; f < 2; ++f) {
          v16h ka = cat8(*(const v8h*)&Ks[kr + f * 32 + hi * 8],
                         *(const v8h*)&Ks[kr + f * 32 + 16 + hi * 8]);
          st[kt] = wmma_f16(ka, qb[f], st[kt]);
        }
      }
      // causal mask + online softmax (per-lane = per-query column)
      const bool needMask = (key0 + 31 > q0w);
      float s0[8], s1[8];
      float rmax = -3.0e38f;
#pragma unroll
      for (int r = 0; r < 8; ++r) {
        const int kg0 = key0 + hi * 8 + r;
        const int kg1 = kg0 + 16;
        s0[r] = (!needMask || kg0 <= qg) ? st[0][r] : -3.0e38f;
        s1[r] = (!needMask || kg1 <= qg) ? st[1][r] : -3.0e38f;
        rmax  = fmaxf(rmax, fmaxf(s0[r], s1[r]));
      }
      rmax = fmaxf(rmax, __shfl_xor(rmax, 16, 32));
      const float mNew  = fmaxf(mrow, rmax);
      const float alpha = __expf(mrow - mNew);
      float rsum = 0.0f;
      v8h p0, p1;
#pragma unroll
      for (int r = 0; r < 8; ++r) {
        const float e0 = __expf(s0[r] - mNew);
        const float e1 = __expf(s1[r] - mNew);
        rsum += e0 + e1;
        p0[r] = (_Float16)e0;
        p1[r] = (_Float16)e1;
      }
      rsum += __shfl_xor(rsum, 16, 32);
      lrow = lrow * alpha + rsum;
      mrow = mNew;
      const v16h pb = cat8(p0, p1);  // P^T as B-operand: in-lane pack

#pragma unroll
      for (int dc = 0; dc < 4; ++dc) {
#pragma unroll
        for (int r = 0; r < 8; ++r) o[dc][r] *= alpha;
        const int dr = (dc * 16 + l) * VT_STRIDE;
        v16h va = cat8(*(const v8h*)&Vt[dr + hi * 8],
                       *(const v8h*)&Vt[dr + 16 + hi * 8]);
        o[dc] = wmma_f16(va, pb, o[dc]);
      }
    }
    __syncthreads();
  }

  // normalize + store ctx as f16 [B, T, D_MODEL]
  const float inv = 1.0f / lrow;
  const int b = bh >> 4;
  const int h = bh & 15;
  const int t = q0w + l;
#pragma unroll
  for (int dc = 0; dc < 4; ++dc) {
    v8h outv;
#pragma unroll
    for (int r = 0; r < 8; ++r) outv[r] = (_Float16)(o[dc][r] * inv);
    *(v8h*)&Ctx[((size_t)(b * SEQ + t)) * D_MODEL + h * HDIM + dc * 16 +
                hi * 8] = outv;
  }
}

// ---------------------------------------------------------------------------
// Kernel 3: output projection  out = ctx @ W_O + b_O (fp32 result)
// Same 128x128x32 pipelined tiling as kernel 1; A is already f16.
// ---------------------------------------------------------------------------
__global__ __launch_bounds__(256)
void out_proj_kernel(const _Float16* __restrict__ Ctx,
                     const float* __restrict__ Wo,
                     const float* __restrict__ bO,
                     float* __restrict__ out) {
  __shared__ _Float16 As[2][128 * AS_STRIDE];
  __shared__ _Float16 Bs[2][128 * BS_STRIDE];

  const int tid  = threadIdx.x;
  const int wave = tid >> 5;
  const int lane = tid & 31;
  const int l    = lane & 15;
  const int hi   = lane >> 4;

  const int m0 = blockIdx.x * 128;
  const int n0 = blockIdx.y * 128;

  const int aRow = tid >> 1;
  const int aK   = (tid & 1) * 16;
  const int bKp  = (tid >> 4) * 2;
  const int bN   = (tid & 15) * 8;

  v8h ra0, ra1;
  v2h rb[8];

  auto gload = [&](int k0) {
    const _Float16* asrc = Ctx + (size_t)(m0 + aRow) * D_MODEL + k0 + aK;
    ra0 = *(const v8h*)(asrc);
    ra1 = *(const v8h*)(asrc + 8);
    const float4* b0 =
        (const float4*)(Wo + (size_t)(k0 + bKp) * D_MODEL + n0 + bN);
    const float4* b1 =
        (const float4*)(Wo + (size_t)(k0 + bKp + 1) * D_MODEL + n0 + bN);
    float4 p0 = b0[0], p1 = b0[1];
    float4 q0 = b1[0], q1 = b1[1];
    float r0[8] = {p0.x, p0.y, p0.z, p0.w, p1.x, p1.y, p1.z, p1.w};
    float r1[8] = {q0.x, q0.y, q0.z, q0.w, q1.x, q1.y, q1.z, q1.w};
#pragma unroll
    for (int j = 0; j < 8; ++j) {
      rb[j][0] = (_Float16)r0[j];
      rb[j][1] = (_Float16)r1[j];
    }
  };
  auto lstore = [&](int buf) {
    *(v8h*)&As[buf][aRow * AS_STRIDE + aK]     = ra0;
    *(v8h*)&As[buf][aRow * AS_STRIDE + aK + 8] = ra1;
#pragma unroll
    for (int j = 0; j < 8; ++j)
      *(v2h*)&Bs[buf][(bN + j) * BS_STRIDE + bKp] = rb[j];
  };

  v8f acc[2][4];
#pragma unroll
  for (int s = 0; s < 2; ++s)
#pragma unroll
    for (int i = 0; i < 4; ++i) acc[s][i] = vzero8();

  const int mb = (wave & 3) * 32;
  const int nb = (wave >> 2) * 64;

  gload(0);
  lstore(0);
  for (int kt = 0; kt < 32; ++kt) {
    const int cur = kt & 1;
    if (kt + 1 < 32) gload((kt + 1) * 32);
    __syncthreads();

    const int r0 = (mb + l) * AS_STRIDE;
    const int r1 = (mb + 16 + l) * AS_STRIDE;
    v16h a0 = cat8(*(const v8h*)&As[cur][r0 + hi * 8],
                   *(const v8h*)&As[cur][r0 + 16 + hi * 8]);
    v16h a1 = cat8(*(const v8h*)&As[cur][r1 + hi * 8],
                   *(const v8h*)&As[cur][r1 + 16 + hi * 8]);
#pragma unroll
    for (int i = 0; i < 4; ++i) {
      const int nr = (nb + i * 16 + l) * BS_STRIDE;
      v16h b = cat8(*(const v8h*)&Bs[cur][nr + hi * 8],
                    *(const v8h*)&Bs[cur][nr + 16 + hi * 8]);
      acc[0][i] = wmma_f16(a0, b, acc[0][i]);
      acc[1][i] = wmma_f16(a1, b, acc[1][i]);
    }
    if (kt + 1 < 32) lstore(cur ^ 1);
  }

#pragma unroll
  for (int s = 0; s < 2; ++s) {
#pragma unroll
    for (int i = 0; i < 4; ++i) {
      const int col    = n0 + nb + i * 16 + l;
      const float bias = bO[col];
#pragma unroll
      for (int r = 0; r < 8; ++r) {
        const int m = m0 + mb + s * 16 + hi * 8 + r;
        out[(size_t)m * D_MODEL + col] = acc[s][i][r] + bias;
      }
    }
  }
}

// ---------------------------------------------------------------------------
extern "C" void kernel_launch(void* const* d_in, const int* in_sizes, int n_in,
                              void* d_out, int out_size, void* d_ws,
                              size_t ws_size, hipStream_t stream) {
  const float* x  = (const float*)d_in[0];
  const float* Wq = (const float*)d_in[1];
  const float* Wk = (const float*)d_in[2];
  const float* Wv = (const float*)d_in[3];
  const float* Wo = (const float*)d_in[4];
  const float* bO = (const float*)d_in[5];
  float* out      = (float*)d_out;

  const size_t elems = (size_t)BATCH * SEQ * D_MODEL;  // 8M f16 each
  _Float16* Qh  = (_Float16*)d_ws;
  _Float16* Kh  = Qh + elems;
  _Float16* Vh  = Kh + elems;
  _Float16* Ctx = Vh + elems;
  (void)in_sizes; (void)n_in; (void)out_size; (void)ws_size;

  // 1) fused QKV projection: M=8192 (64 tiles), N=3072 (24 tiles of 128)
  qkv_proj_kernel<<<dim3(64, 24), 256, 0, stream>>>(x, Wq, Wk, Wv, Qh, Kh, Vh);

  // 2) causal flash attention: 32 query tiles x 64 (b,h) slabs
  attention_kernel<<<dim3(32, 64), 128, 0, stream>>>(Qh, Kh, Vh, Ctx);

  // 3) output projection: M=8192 (64 tiles), N=1024 (8 tiles of 128)
  out_proj_kernel<<<dim3(64, 8), 256, 0, stream>>>(Ctx, Wo, bO, out);
}
